// CViT_4114578670199
// MI455X (gfx1250) — matrix operand order repaired and checked by reading, not compiled, not run on hardware
//
#include <hip/hip_runtime.h>
#include <math.h>

// ---------------------------------------------------------------------------
// CDNA5 / gfx1250 ViT encoder forward.  Heavy math on v_wmma_f32_16x16x32_bf16
// (f32 accumulate); LN / softmax / GELU / residual in fp32.  Compute-bound
// (~360 GFLOP vs ~300MB weights @ 23.3 TB/s).  Weights are converted
// fp32->bf16 AND transposed to [N][K] once, so GEMM tiles are contiguous 16B
// global->LDS copies done with GLOBAL_LOAD_ASYNC_TO_LDS_B128 (ASYNCcnt),
// double-buffered so fills fully overlap the WMMA stage.
// ---------------------------------------------------------------------------

#define USE_ASYNC_LDS 1   // verified: assembler accepts the mnemonic (round 2)

typedef __bf16 bf16_t;
typedef __attribute__((ext_vector_type(16))) __bf16 v16bf;
typedef __attribute__((ext_vector_type(8)))  float  v8f;

union FragBF { v16bf v; uint4 q[2]; bf16_t e[16]; };

static __device__ __forceinline__ v8f wmma_bf16(const FragBF& a, const FragBF& b, v8f c) {
  return __builtin_amdgcn_wmma_f32_16x16x32_bf16(false, a.v, false, b.v,
                                                 (short)0, c, false, false);
}

static __device__ __forceinline__ float gelu_exact(float x) {
  return 0.5f * x * (1.0f + erff(x * 0.70710678118654752f));
}

// 16B global -> LDS copy.  Async path: GLOBAL_LOAD_ASYNC_TO_LDS_B128
// (VDST = VGPR with LDS byte offset, VADDR = 64-bit global address, SADDR=off).
static __device__ __forceinline__ void copy16_g2l(void* lds, const void* g) {
#if USE_ASYNC_LDS
  unsigned l = (unsigned)(uintptr_t)lds;
  unsigned long long a = (unsigned long long)(uintptr_t)g;
  asm volatile("global_load_async_to_lds_b128 %0, %1, off"
               :: "v"(l), "v"(a) : "memory");
#else
  *(uint4*)lds = *(const uint4*)g;
#endif
}

// Wait until at most 8 async ops outstanding (previous stage's 8 fills done;
// async loads retire in order) / until all done.
static __device__ __forceinline__ void wait_async_le8() {
#if USE_ASYNC_LDS
  asm volatile("s_wait_asynccnt 0x8" ::: "memory");
#endif
}
static __device__ __forceinline__ void wait_async_le0() {
#if USE_ASYNC_LDS
  asm volatile("s_wait_asynccnt 0x0" ::: "memory");
#endif
}

// ---------------------------------------------------------------------------
// Weight convert + transpose: W fp32 [K][N]  ->  Wt bf16 [N][K].
// 64x64 tiles through LDS; both global phases coalesced.  blockIdx.z = layer.
// ---------------------------------------------------------------------------
__global__ __launch_bounds__(256)
void cvit_tcvt(const float* __restrict__ W, bf16_t* __restrict__ Wt,
               int K, int N) {
  __shared__ bf16_t Ls[64][65];
  const size_t layer = (size_t)blockIdx.z * K * N;
  const float* Wl  = W  + layer;
  bf16_t*      Wtl = Wt + layer;
  const int n0 = blockIdx.x * 64;
  const int k0 = blockIdx.y * 64;
  const int t = threadIdx.x;
#pragma unroll
  for (int p = 0; p < 16; ++p) {
    int id = t + p * 256;
    int r = id >> 6, c = id & 63;
    Ls[r][c] = (bf16_t)Wl[(size_t)(k0 + r) * N + n0 + c];
  }
  __syncthreads();
#pragma unroll
  for (int p = 0; p < 16; ++p) {
    int id = t + p * 256;
    int rr = id >> 6, cc = id & 63;                 // rr: n-local, cc: k-local
    Wtl[(size_t)(n0 + rr) * K + k0 + cc] = Ls[cc][rr];
  }
}

// ---------------------------------------------------------------------------
// LayerNorm over D=1024, write bf16.  One block (256 threads) per row.
// ---------------------------------------------------------------------------
__global__ __launch_bounds__(256)
void cvit_ln(const float* __restrict__ x, const float* __restrict__ s,
             const float* __restrict__ b, bf16_t* __restrict__ out) {
  const int row = blockIdx.x;
  const int t = threadIdx.x;
  const float* xr = x + (size_t)row * 1024;
  float v[4], sum = 0.f, sq = 0.f;
#pragma unroll
  for (int i = 0; i < 4; ++i) {
    v[i] = xr[t + 256 * i];
    sum += v[i];
    sq  += v[i] * v[i];
  }
  __shared__ float s1[256], s2[256];
  s1[t] = sum; s2[t] = sq;
  __syncthreads();
  for (int o = 128; o > 0; o >>= 1) {
    if (t < o) { s1[t] += s1[t + o]; s2[t] += s2[t + o]; }
    __syncthreads();
  }
  const float mean = s1[0] * (1.0f / 1024.0f);
  const float var  = s2[0] * (1.0f / 1024.0f) - mean * mean;
  const float rstd = rsqrtf(var + 1e-5f);
#pragma unroll
  for (int i = 0; i < 4; ++i) {
    int c = t + 256 * i;
    out[(size_t)row * 1024 + c] = (bf16_t)((v[i] - mean) * rstd * s[c] + b[c]);
  }
}

// ---------------------------------------------------------------------------
// Repack V: qkv[b*1024+n][2048 + h*64 + d]  ->  vT[((b*16+h)*64+d)*1024 + n]
// ---------------------------------------------------------------------------
__global__ void cvit_repack_v(const bf16_t* __restrict__ qkv,
                              bf16_t* __restrict__ vT, int total) {
  int i = blockIdx.x * blockDim.x + threadIdx.x;
  if (i >= total) return;
  int n = i & 1023;
  int d = (i >> 10) & 63;
  int h = (i >> 16) & 15;
  int b = i >> 20;
  vT[i] = qkv[(size_t)(b * 1024 + n) * 3072 + 2048 + h * 64 + d];
}

// ---------------------------------------------------------------------------
// bf16 WMMA GEMM:  C[M,N] = A[M,K] * Bt[N,K]^T   (B pre-transposed, [N][K])
// Block tile 128x128, BK=64, 256 threads = 8 waves (4x2), wave tile 32x64
// (2x4 sub-tiles), 16 WMMA per LDS stage.  Double-buffered async LDS fills:
// stage s+1's 8 async b128 copies per wave are in flight while stage s's
// WMMAs run; s_wait_asynccnt 8 retires exactly the previous stage (in-order).
// MODE 0: store bf16            (QKV)
// MODE 1: +bias, GELU, bf16     (MLP up)
// MODE 2: +bias +residual, f32  (out-proj / MLP down)
// ---------------------------------------------------------------------------
template <int MODE>
__global__ __launch_bounds__(256)
void cvit_gemm(const bf16_t* __restrict__ A, const bf16_t* __restrict__ Bt,
               const float* __restrict__ bias, const float* __restrict__ resid,
               void* __restrict__ dstv, int M, int N, int K) {
  __shared__ __align__(16) bf16_t As[2][128][72];   // 64 + 8 pad, rows 16B aligned
  __shared__ __align__(16) bf16_t Bs[2][128][72];   // Bs[buf][col][k]

  const int tid  = threadIdx.x;
  const int lane = tid & 31;
  const int wave = tid >> 5;
  const int hf   = lane >> 4;     // lane half
  const int ln   = lane & 15;
  const int wm   = wave >> 1;     // 0..3
  const int wn   = wave & 1;      // 0..1
  const int m0   = blockIdx.y * 128;
  const int n0   = blockIdx.x * 128;

  // Tile-fill assignments: 1024 16B chunks per 128x64 tile, 4 per thread.
  const int frow = tid >> 3;            // 0..31 (advance by 32 rows per pass)
  const int fcol = (tid & 7) * 8;       // 0..56 (k-local, elems)

  auto issue_stage = [&](int buf, int kk) {
#pragma unroll
    for (int p = 0; p < 4; ++p) {
      int r = frow + p * 32;
      copy16_g2l(&As[buf][r][fcol], A  + (size_t)(m0 + r) * K + kk + fcol);
      copy16_g2l(&Bs[buf][r][fcol], Bt + (size_t)(n0 + r) * K + kk + fcol);
    }
  };

  v8f acc[2][4];
#pragma unroll
  for (int mi = 0; mi < 2; ++mi)
#pragma unroll
    for (int ni = 0; ni < 4; ++ni)
      acc[mi][ni] = (v8f){0.f, 0.f, 0.f, 0.f, 0.f, 0.f, 0.f, 0.f};

  const int S = K >> 6;            // stages of BK=64
  issue_stage(0, 0);

  for (int s = 0; s < S; ++s) {
    const int cur = s & 1;
    if (s + 1 < S) {
      issue_stage(cur ^ 1, (s + 1) * 64);   // prefetch next stage
      wait_async_le8();                     // previous 8 (this stage) retired
    } else {
      wait_async_le0();
    }
    __syncthreads();                        // all waves' fills visible

#pragma unroll
    for (int ks = 0; ks < 2; ++ks) {
      FragBF af[2], bfr[4];
#pragma unroll
      for (int mi = 0; mi < 2; ++mi) {
        int mr = wm * 32 + mi * 16 + ln;
        int ko = ks * 32 + hf * 8;           // A halves: K 0-7 / 8-15 (+16)
        af[mi].q[0] = *(const uint4*)&As[cur][mr][ko];
        af[mi].q[1] = *(const uint4*)&As[cur][mr][ko + 16];
      }
#pragma unroll
      for (int ni = 0; ni < 4; ++ni) {
        int nc = wn * 64 + ni * 16 + ln;
        int ko = ks * 32 + hf * 8;           // B halves: K 0-7 / 8-15 (+16)
        bfr[ni].q[0] = *(const uint4*)&Bs[cur][nc][ko];
        bfr[ni].q[1] = *(const uint4*)&Bs[cur][nc][ko + 16];
      }
#pragma unroll
      for (int mi = 0; mi < 2; ++mi)
#pragma unroll
        for (int ni = 0; ni < 4; ++ni)
          acc[mi][ni] = wmma_bf16(af[mi], bfr[ni], acc[mi][ni]);
    }
    __syncthreads();   // all waves done reading buf `cur` before stage s+2 refills it
  }

  // Epilogue.  C layout: VGPR r -> M = r (lanes 0-15) / r+8 (lanes 16-31), N = ln.
#pragma unroll
  for (int mi = 0; mi < 2; ++mi)
#pragma unroll
    for (int ni = 0; ni < 4; ++ni) {
      const int col = n0 + wn * 64 + ni * 16 + ln;
#pragma unroll
      for (int r = 0; r < 8; ++r) {
        const int row = m0 + wm * 32 + mi * 16 + hf * 8 + r;
        float v = acc[mi][ni][r];
        if (MODE == 0) {
          ((bf16_t*)dstv)[(size_t)row * N + col] = (bf16_t)v;
        } else if (MODE == 1) {
          v += bias[col];
          ((bf16_t*)dstv)[(size_t)row * N + col] = (bf16_t)gelu_exact(v);
        } else {
          v += bias[col] + resid[(size_t)row * N + col];
          ((float*)dstv)[(size_t)row * N + col] = v;
        }
      }
    }
}

// ---------------------------------------------------------------------------
// Flash attention. Block = (b, h, 64-query tile), 128 threads = 4 waves,
// each wave owns 16 query rows.  Softmax scale 1/32 (=2^-5, exact in bf16)
// folded into Q fragments.  S = Q K^T, online softmax, O += P V.
// ---------------------------------------------------------------------------
__global__ __launch_bounds__(128)
void cvit_attn(const bf16_t* __restrict__ qkv, const bf16_t* __restrict__ vT,
               bf16_t* __restrict__ out) {
  const int bid = blockIdx.x;
  const int qt  = bid & 15;
  const int h   = (bid >> 4) & 15;
  const int b   = bid >> 8;
  const int tid = threadIdx.x;
  const int wave = tid >> 5;
  const int lane = tid & 31;
  const int hf   = lane >> 4;
  const int ln   = lane & 15;
  const int q0   = qt * 64 + wave * 16;

  __shared__ __align__(16) bf16_t Pl[4][16][72];   // per-wave P scratch (+pad)

  // Q fragments (A operand), pre-scaled by dim^-0.5 = 1/32
  FragBF aq[2];
  const size_t rowQ = (size_t)(b * 1024 + q0 + ln) * 3072;
#pragma unroll
  for (int ks = 0; ks < 2; ++ks) {
    const bf16_t* p = qkv + rowQ + h * 64 + ks * 32 + hf * 8;
    aq[ks].q[0] = *(const uint4*)p;
    aq[ks].q[1] = *(const uint4*)(p + 16);
#pragma unroll
    for (int i = 0; i < 16; ++i)
      aq[ks].e[i] = (bf16_t)((float)aq[ks].e[i] * 0.03125f);
  }

  float rm[8], rs[8];
#pragma unroll
  for (int r = 0; r < 8; ++r) { rm[r] = -3.0e38f; rs[r] = 0.f; }
  v8f ao[4];
#pragma unroll
  for (int t = 0; t < 4; ++t)
    ao[t] = (v8f){0.f, 0.f, 0.f, 0.f, 0.f, 0.f, 0.f, 0.f};

  for (int j0 = 0; j0 < 1024; j0 += 64) {
    // S tiles: Q (16x64) x K^T (64 x 64-key chunk)
    v8f st[4];
#pragma unroll
    for (int t = 0; t < 4; ++t) {
      st[t] = (v8f){0.f, 0.f, 0.f, 0.f, 0.f, 0.f, 0.f, 0.f};
      const bf16_t* kr =
          qkv + (size_t)(b * 1024 + j0 + t * 16 + ln) * 3072 + 1024 + h * 64;
      FragBF kb;
#pragma unroll
      for (int ks = 0; ks < 2; ++ks) {
        const bf16_t* p = kr + ks * 32 + hf * 8;
        kb.q[0] = *(const uint4*)p;
        kb.q[1] = *(const uint4*)(p + 16);
        st[t] = wmma_bf16(aq[ks], kb, st[t]);
      }
    }

    // Online softmax (rows striped: VGPR r -> M = r + hf*8; reduce over the
    // 16 lanes of the same half with xor masks 1/2/4/8).
    float cm[8];
#pragma unroll
    for (int r = 0; r < 8; ++r) {
      float m = st[0][r];
#pragma unroll
      for (int t = 1; t < 4; ++t) m = fmaxf(m, st[t][r]);
#pragma unroll
      for (int o = 1; o < 16; o <<= 1) m = fmaxf(m, __shfl_xor(m, o, 32));
      cm[r] = m;
    }
    float al[8];
#pragma unroll
    for (int r = 0; r < 8; ++r) {
      float nm = fmaxf(rm[r], cm[r]);
      al[r] = __expf(rm[r] - nm);
      rm[r] = nm;
      rs[r] *= al[r];
    }
#pragma unroll
    for (int t = 0; t < 4; ++t)
#pragma unroll
      for (int r = 0; r < 8; ++r) ao[t][r] *= al[r];

    float ps[8];
#pragma unroll
    for (int r = 0; r < 8; ++r) ps[r] = 0.f;
#pragma unroll
    for (int t = 0; t < 4; ++t)
#pragma unroll
      for (int r = 0; r < 8; ++r) {
        float p = __expf(st[t][r] - rm[r]);
        ps[r] += p;
        Pl[wave][hf * 8 + r][t * 16 + ln] = (bf16_t)p;   // C-layout -> LDS
      }
#pragma unroll
    for (int r = 0; r < 8; ++r) {
      float s = ps[r];
#pragma unroll
      for (int o = 1; o < 16; o <<= 1) s += __shfl_xor(s, o, 32);
      rs[r] += s;
    }

    // P fragments (A operand) from per-wave LDS; V fragments from vT (contig).
    FragBF ap[2];
#pragma unroll
    for (int ks = 0; ks < 2; ++ks) {
      const bf16_t* p = &Pl[wave][ln][ks * 32 + hf * 8];
      ap[ks].q[0] = *(const uint4*)p;
      ap[ks].q[1] = *(const uint4*)(p + 16);
    }
#pragma unroll
    for (int t2 = 0; t2 < 4; ++t2) {
      const bf16_t* vr =
          vT + (size_t)((b * 16 + h) * 64 + t2 * 16 + ln) * 1024 + j0;
      FragBF bv;
#pragma unroll
      for (int ks = 0; ks < 2; ++ks) {
        const bf16_t* p = vr + ks * 32 + hf * 8;
        bv.q[0] = *(const uint4*)p;
        bv.q[1] = *(const uint4*)(p + 16);
        ao[t2] = wmma_bf16(ap[ks], bv, ao[t2]);
      }
    }
  }

  // O = ao / rowsum  ->  attn_out[b*1024+q][h*64+d]  (bf16)
#pragma unroll
  for (int t = 0; t < 4; ++t)
#pragma unroll
    for (int r = 0; r < 8; ++r) {
      float o = ao[t][r] / rs[r];
      int row = b * 1024 + q0 + hf * 8 + r;
      out[(size_t)row * 1024 + h * 64 + t * 16 + ln] = (bf16_t)o;
    }
}

// ---------------------------------------------------------------------------
// Host: orchestrate 6 layers.  d_out doubles as the fp32 residual stream x.
// ---------------------------------------------------------------------------
extern "C" void kernel_launch(void* const* d_in, const int* in_sizes, int n_in,
                              void* d_out, int out_size, void* d_ws, size_t ws_size,
                              hipStream_t stream) {
  const float* x      = (const float*)d_in[0];
  const float* ln1_s  = (const float*)d_in[1];
  const float* ln1_b  = (const float*)d_in[2];
  const float* w_qkv  = (const float*)d_in[3];
  const float* w_out  = (const float*)d_in[4];
  const float* b_out  = (const float*)d_in[5];
  const float* ln2_s  = (const float*)d_in[6];
  const float* ln2_b  = (const float*)d_in[7];
  const float* w1     = (const float*)d_in[8];
  const float* b1     = (const float*)d_in[9];
  const float* w2     = (const float*)d_in[10];
  const float* b2     = (const float*)d_in[11];

  const int D = 1024, MLP = 4096, DEPTH = 6;
  const int B = in_sizes[0] / (1024 * 1024);   // 2
  const int M = B * 1024;                      // rows (B*N)

  // Workspace carve (bf16 transposed weights + activation scratch).
  char* ws = (char*)d_ws;
  size_t off = 0;
  auto carve = [&](size_t bytes) { char* p = ws + off; off += (bytes + 255) & ~(size_t)255; return p; };
  bf16_t* wqkv_t = (bf16_t*)carve((size_t)DEPTH * D * 3 * D * 2);   // [3D][D] per layer
  bf16_t* wout_t = (bf16_t*)carve((size_t)DEPTH * D * D * 2);       // [D][D]
  bf16_t* w1_t   = (bf16_t*)carve((size_t)DEPTH * D * MLP * 2);     // [MLP][D]
  bf16_t* w2_t   = (bf16_t*)carve((size_t)DEPTH * MLP * D * 2);     // [D][MLP]
  bf16_t* h_b    = (bf16_t*)carve((size_t)M * D * 2);
  bf16_t* qkv_b  = (bf16_t*)carve((size_t)M * 3 * D * 2);
  bf16_t* vT_b   = (bf16_t*)carve((size_t)M * D * 2);
  bf16_t* ao_b   = (bf16_t*)carve((size_t)M * D * 2);
  bf16_t* a1_b   = (bf16_t*)carve((size_t)M * MLP * 2);
  (void)ws_size; (void)n_in; (void)out_size;

  float* xbuf = (float*)d_out;
  hipMemcpyAsync(xbuf, x, (size_t)M * D * 4, hipMemcpyDeviceToDevice, stream);

  // One-shot convert + transpose of all weights (deterministic, every call).
  cvit_tcvt<<<dim3(3 * D / 64, D / 64, DEPTH), 256, 0, stream>>>(w_qkv, wqkv_t, D, 3 * D);
  cvit_tcvt<<<dim3(D / 64, D / 64, DEPTH),     256, 0, stream>>>(w_out, wout_t, D, D);
  cvit_tcvt<<<dim3(MLP / 64, D / 64, DEPTH),   256, 0, stream>>>(w1,    w1_t,   D, MLP);
  cvit_tcvt<<<dim3(D / 64, MLP / 64, DEPTH),   256, 0, stream>>>(w2,    w2_t,   MLP, D);

  for (int l = 0; l < DEPTH; ++l) {
    const bf16_t* Wq = wqkv_t + (size_t)l * D * 3 * D;
    const bf16_t* Wo = wout_t + (size_t)l * D * D;
    const bf16_t* W1 = w1_t   + (size_t)l * D * MLP;
    const bf16_t* W2 = w2_t   + (size_t)l * MLP * D;

    // LN1 -> h (bf16)
    cvit_ln<<<M, 256, 0, stream>>>(xbuf, ln1_s + l * D, ln1_b + l * D, h_b);
    // QKV GEMM -> qkv (bf16)
    cvit_gemm<0><<<dim3(3 * D / 128, M / 128), 256, 0, stream>>>(
        h_b, Wq, nullptr, nullptr, qkv_b, M, 3 * D, D);
    // V transpose
    cvit_repack_v<<<(M * D + 255) / 256, 256, 0, stream>>>(qkv_b, vT_b, M * D);
    // Flash attention -> ao (bf16)
    cvit_attn<<<B * 256, 128, 0, stream>>>(qkv_b, vT_b, ao_b);
    // Out-proj + bias + residual -> x (fp32)
    cvit_gemm<2><<<dim3(D / 128, M / 128), 256, 0, stream>>>(
        ao_b, Wo, b_out + l * D, xbuf, xbuf, M, D, D);
    // LN2 -> h (bf16)
    cvit_ln<<<M, 256, 0, stream>>>(xbuf, ln2_s + l * D, ln2_b + l * D, h_b);
    // MLP up + GELU -> a1 (bf16)
    cvit_gemm<1><<<dim3(MLP / 128, M / 128), 256, 0, stream>>>(
        h_b, W1, b1 + l * MLP, nullptr, a1_b, M, MLP, D);
    // MLP down + bias + residual -> x (fp32)
    cvit_gemm<2><<<dim3(D / 128, M / 128), 256, 0, stream>>>(
        a1_b, W2, b2 + l * D, xbuf, xbuf, M, D, MLP);
  }
}
